// dMaSIF_19988777795865
// MI455X (gfx1250) — compile-verified, matrix operand order checked by reading, not run on hardware
//
#include <hip/hip_runtime.h>
#include <hip/hip_bf16.h>

// ---------------------------------------------------------------------------
// dMaSIF-style pipeline for MI455X (gfx1250, wave32, WMMA)
//   - message-passing GEMMs via v_wmma_f32_16x16x32_bf16 (f32 accumulate)
//   - one workgroup (4 wave32) per point: 16 neighbor rows = one WMMA M-tile
//   - bf16 mirrors of feature tables so tile staging is a raw byte copy,
//     routed through global_load_async_to_lds_b128 (ASYNCcnt) when available
//   - chunk-parallel KNN (8 threads/query) to expose enough parallelism
// ---------------------------------------------------------------------------

#define DEV static __device__ __forceinline__

typedef __attribute__((ext_vector_type(16))) __bf16 v16bf;
typedef __attribute__((ext_vector_type(8)))  float  v8f;
typedef __attribute__((ext_vector_type(4)))  int    v4i;

union FragBF {
    uint4 u[2];
    v16bf v;
};

#if defined(__gfx1250__) && \
    __has_builtin(__builtin_amdgcn_global_load_async_to_lds_b128) && \
    __has_builtin(__builtin_amdgcn_s_wait_asynccnt)
#define HAVE_ASYNC_LDS 1
#else
#define HAVE_ASYNC_LDS 0
#endif

#if HAVE_ASYNC_LDS
// Cross-address-space casts routed through integers (clang forbids direct
// pointer casts between address spaces). Generic->AS1 keeps the 64-bit
// address; generic->AS3 truncates to the low 32 bits, which per the CDNA5
// aperture rule is exactly the LDS offset.
DEV __attribute__((address_space(1))) v4i* as_global_v4i(const void* p) {
    return (__attribute__((address_space(1))) v4i*)(unsigned long long)(size_t)p;
}
DEV __attribute__((address_space(3))) v4i* as_lds_v4i(void* p) {
    return (__attribute__((address_space(3))) v4i*)(unsigned)(size_t)p;
}
#endif

DEV unsigned short f32_to_bf16(float f) {
    union { float f; unsigned u; } x; x.f = f;
    unsigned r = x.u + 0x7FFFu + ((x.u >> 16) & 1u);   // round-to-nearest-even
    return (unsigned short)(r >> 16);
}
DEV float bf16_to_f32(unsigned short h) {
    union { unsigned u; float f; } x; x.u = ((unsigned)h) << 16;
    return x.f;
}
DEV float leakyf(float x) { return x > 0.f ? x : 0.2f * x; }

// ---------------------------------------------------------------------------
// Weight fragment prep: fp32 [NL][K(H rows)][ncols] -> bf16 fragments laid out
// [layer][tile][chunk][lane][16] matching the wave32 16-bit B-matrix layout:
//   lane n (<16):  col n, K = chunk*32 + {0..7, 16..23}
//   lane n+16:     col n, K = chunk*32 + {8..15, 24..31}
// Only K < 256 goes through WMMA; K == 256 is applied as a rank-1 VALU term.
// ---------------------------------------------------------------------------
__global__ void prep_wfrag(const float* __restrict__ src,
                           unsigned short* __restrict__ dst,
                           int ntiles, int rowstride, int nlayers,
                           int layer_src_elems) {
    int total = nlayers * ntiles * 8 * 32 * 16;
    for (int t = blockIdx.x * blockDim.x + threadIdx.x; t < total;
         t += gridDim.x * blockDim.x) {
        int j     = t & 15;
        int lane  = (t >> 4) & 31;
        int chunk = (t >> 9) & 7;
        int rest  = t >> 12;
        int tile  = rest % ntiles;
        int layer = rest / ntiles;
        int klocal = (j & 7) + ((j >> 3) << 4) + ((lane >= 16) ? 8 : 0);
        int K = chunk * 32 + klocal;
        int n = tile * 16 + (lane & 15);
        dst[t] = f32_to_bf16(src[(size_t)layer * layer_src_elems +
                                 (size_t)K * rowstride + n]);
    }
}

// ---------------------------------------------------------------------------
// Atom-type embedding: t = leaky(at @ w1 + b1) @ w2 + b2   (M x 128 + mirror)
// ---------------------------------------------------------------------------
__global__ void atom_embed(const float* __restrict__ at,
                           const float* __restrict__ w1, const float* __restrict__ b1,
                           const float* __restrict__ w2, const float* __restrict__ b2,
                           float* __restrict__ outp, unsigned short* __restrict__ outb,
                           int M_) {
    int m = blockIdx.x * blockDim.x + threadIdx.x;
    if (m >= M_) return;
    float a[6], h[6];
#pragma unroll
    for (int c = 0; c < 6; ++c) a[c] = at[m * 6 + c];
#pragma unroll
    for (int c = 0; c < 6; ++c) {
        float s = b1[c];
#pragma unroll
        for (int r = 0; r < 6; ++r) s += a[r] * w1[r * 6 + c];
        h[c] = leakyf(s);
    }
    for (int d = 0; d < 128; ++d) {
        float s = b2[d];
#pragma unroll
        for (int c = 0; c < 6; ++c) s += h[c] * w2[c * 128 + d];
        outp[(size_t)m * 128 + d] = s;
        outb[(size_t)m * 128 + d] = f32_to_bf16(s);
    }
}

__global__ void fill_ones_pair(float* __restrict__ p, unsigned short* __restrict__ b,
                               int n) {
    for (int i = blockIdx.x * blockDim.x + threadIdx.x; i < n;
         i += gridDim.x * blockDim.x) {
        p[i] = 1.0f;
        b[i] = 0x3F80;   // bf16(1.0)
    }
}

// ---------------------------------------------------------------------------
// Chunk-parallel brute-force KNN.
// 128-thread blocks: 16 queries x 8 candidate chunks. Each sub-thread keeps a
// sorted top-KK list over its C/8 candidates; the 8 sorted lists per query
// are merged through LDS (ties -> lower index). DROP=1 drops the self
// neighbor (atom->atom case).
// ---------------------------------------------------------------------------
template <int KK, int DROP>
__global__ void __launch_bounds__(128)
knn_kernel(const float* __restrict__ qxyz, const float* __restrict__ cxyz,
           const int* __restrict__ qb, const int* __restrict__ cb,
           int Q, int C, int* __restrict__ oidx, float* __restrict__ odist) {
    __shared__ float sbd[16][8][KK];
    __shared__ int   sbi[16][8][KK];

    const int ql = threadIdx.x >> 3;     // query slot within block
    const int ch = threadIdx.x & 7;      // candidate chunk
    const int q  = blockIdx.x * 16 + ql;
    const bool valid = (q < Q);

    float bd[KK];
    int   bi[KK];
#pragma unroll
    for (int i = 0; i < KK; ++i) { bd[i] = __builtin_inff(); bi[i] = 0; }

    if (valid) {
        const float qx = qxyz[q * 3], qy = qxyz[q * 3 + 1], qz = qxyz[q * 3 + 2];
        const int qbv = qb[q];
        const int j0 = (C * ch) >> 3;
        const int j1 = (C * (ch + 1)) >> 3;
        for (int j = j0; j < j1; ++j) {
            float dx = qx - cxyz[j * 3];
            float dy = qy - cxyz[j * 3 + 1];
            float dz = qz - cxyz[j * 3 + 2];
            float d2 = dx * dx + dy * dy + dz * dz;
            if (cb[j] != qbv) d2 = __builtin_inff();
            if (d2 < bd[KK - 1]) {
                int pos = KK - 1;
                while (pos > 0 && bd[pos - 1] > d2) {
                    bd[pos] = bd[pos - 1];
                    bi[pos] = bi[pos - 1];
                    --pos;
                }
                bd[pos] = d2;
                bi[pos] = j;
            }
        }
    }
#pragma unroll
    for (int i = 0; i < KK; ++i) { sbd[ql][ch][i] = bd[i]; sbi[ql][ch][i] = bi[i]; }
    __syncthreads();

    if (valid && ch == 0) {
        float fd[KK];
        int   fi[KK];
#pragma unroll
        for (int i = 0; i < KK; ++i) { fd[i] = __builtin_inff(); fi[i] = 0x7FFFFFFF; }
        for (int c = 0; c < 8; ++c) {
            for (int s = 0; s < KK; ++s) {
                float d = sbd[ql][c][s];
                int   ix = sbi[ql][c][s];
                if (d < fd[KK - 1] ||
                    (d == fd[KK - 1] && ix < fi[KK - 1])) {
                    int pos = KK - 1;
                    while (pos > 0 && (fd[pos - 1] > d ||
                                       (fd[pos - 1] == d && fi[pos - 1] > ix))) {
                        fd[pos] = fd[pos - 1];
                        fi[pos] = fi[pos - 1];
                        --pos;
                    }
                    fd[pos] = d;
                    fi[pos] = ix;
                } else {
                    break;   // sorted source list: rest can't qualify either
                }
            }
        }
#pragma unroll
        for (int s = 0; s < 16; ++s) {
            oidx[(size_t)q * 16 + s]  = fi[s + DROP];
            odist[(size_t)q * 16 + s] = fd[s + DROP];
        }
    }
}

// ---------------------------------------------------------------------------
// One message-passing layer, one workgroup (128 threads = 4 wave32) per point:
//   f (16 x 257)  = [center bcast | nbr gather | d2]
//   h = leaky(f @ w1 + b1)            first GEMM: 16 N-tiles x 8 K-chunks WMMA
//   msg = h @ w2 + b2                 second GEMM: 8 N-tiles x 8 K-chunks WMMA
//   out = center + leaky(groupnorm(sum_k msg))
// Tile staging copies bf16 mirror rows straight into LDS (async when
// available: global_load_async_to_lds_b128 + s_wait_asynccnt).
// ---------------------------------------------------------------------------
__global__ void __launch_bounds__(128)
mp_layer(int P,
         const float* __restrict__ center,          // P x 128 fp32 (residual)
         const unsigned short* __restrict__ centb,  // P x 128 bf16 mirror
         const unsigned short* __restrict__ nbrb,   // neighbor table bf16
         const int*   __restrict__ nidx,            // P x 16
         const float* __restrict__ ndist,           // P x 16 (squared dists)
         const uint4* __restrict__ w1frag,          // bf16 frags (layer offset)
         const uint4* __restrict__ w2frag,
         const float* __restrict__ w1fp,            // fp32 257x257 (layer offs)
         const float* __restrict__ w2fp,            // fp32 257x128
         const float* __restrict__ b1,              // 257
         const float* __restrict__ b2,              // 128
         const float* __restrict__ gnw, const float* __restrict__ gnb,  // 128
         float* __restrict__ outp,                  // P x 128 fp32
         unsigned short* __restrict__ outb)         // P x 128 bf16 mirror
{
    constexpr int H = 257;
    __shared__ __align__(16) unsigned short fA[16 * 256];  // bf16 feature tile
    __shared__ __align__(16) unsigned short hB[16 * 256];  // bf16 hidden tile
    __shared__ int   sidx[16];
    __shared__ float dcol[16];     // distance column (f[:,256])
    __shared__ float hcol[16];     // hidden column  (h[:,256])
    __shared__ float msum[8 * 32]; // reused: hcol partials, then GEMM2 sums
    __shared__ float colsum[128];
    __shared__ float gstat[4];     // mu0, rstd0, mu1, rstd1

    const int p    = blockIdx.x;
    const int tid  = threadIdx.x;
    const int lane = tid & 31;
    const int wave = tid >> 5;

    if (tid < 16) {
        sidx[tid] = nidx[(size_t)p * 16 + tid];
        dcol[tid] = ndist[(size_t)p * 16 + tid];
    }
    __syncthreads();

    // ---- stage f tile: 16 rows x 512B, 16B per async op, 4 ops per thread
    for (int s = tid; s < 512; s += 128) {
        int r   = s >> 5;          // row
        int seg = s & 31;          // 16B segment within row
        const unsigned short* src =
            (seg < 16) ? (centb + (size_t)p * 128 + seg * 8)
                       : (nbrb + (size_t)sidx[r] * 128 + (seg - 16) * 8);
        unsigned short* dst = fA + r * 256 + seg * 8;
#if HAVE_ASYNC_LDS
        __builtin_amdgcn_global_load_async_to_lds_b128(
            as_global_v4i(src), as_lds_v4i(dst), 0, 0);
#else
        *(uint4*)dst = *(const uint4*)src;
#endif
    }
#if HAVE_ASYNC_LDS
    __builtin_amdgcn_s_wait_asynccnt(0);
#endif
    __syncthreads();

    const int arow_off = (lane >= 16) ? 8 : 0;
    const int arow     = lane & 15;
    const int ncol     = lane & 15;   // column within N-tile for this lane

    // ---- GEMM 1: h[:,0:256] = leaky(f @ w1 + b1), 16 N-tiles across 4 waves
    for (int tile = wave * 4; tile < wave * 4 + 4; ++tile) {
        float bn = b1[tile * 16 + ncol];
        v8f acc;
#pragma unroll
        for (int r = 0; r < 8; ++r) acc[r] = bn;
#pragma unroll
        for (int chunk = 0; chunk < 8; ++chunk) {
            FragBF a, b;
            const uint4* ap =
                (const uint4*)(fA + arow * 256 + chunk * 32 + arow_off);
            a.u[0] = ap[0];
            a.u[1] = ap[2];   // +16 bf16 elements
            int base = (((tile * 8 + chunk) * 32) + lane) << 1;
            b.u[0] = w1frag[base];
            b.u[1] = w1frag[base + 1];
            acc = __builtin_amdgcn_wmma_f32_16x16x32_bf16(
                false, a.v, false, b.v, (short)0, acc, false, false);
        }
        // rank-1 term for K=256 (distance column), then activation
        float wc = w1fp[(size_t)256 * H + tile * 16 + ncol];
#pragma unroll
        for (int r = 0; r < 8; ++r) {
            int row = (lane < 16) ? r : (r + 8);
            float v = leakyf(acc[r] + dcol[row] * wc);
            hB[row * 256 + tile * 16 + ncol] = f32_to_bf16(v);
        }
    }

    // ---- hidden column 256 (dot against w1[:,256]); parallel partials
    {
        int row = tid >> 3, kb = tid & 7;
        float s = 0.f;
        for (int k = kb * 32; k < kb * 32 + 32; ++k)
            s += bf16_to_f32(fA[row * 256 + k]) * w1fp[(size_t)k * H + 256];
        __syncthreads();          // also covers GEMM1 hB stores
        msum[tid] = s;
    }
    __syncthreads();
    if (tid < 16) {
        float s = b1[256] + dcol[tid] * w1fp[(size_t)256 * H + 256];
#pragma unroll
        for (int c = 0; c < 8; ++c) s += msum[tid * 8 + c];
        hcol[tid] = leakyf(s);
    }
    __syncthreads();

    // ---- GEMM 2: msg = h @ w2 + b2 (8 N-tiles), fused neighbor-sum
    for (int tile = wave * 2; tile < wave * 2 + 2; ++tile) {
        float bn = b2[tile * 16 + ncol];
        v8f acc;
#pragma unroll
        for (int r = 0; r < 8; ++r) acc[r] = bn;
#pragma unroll
        for (int chunk = 0; chunk < 8; ++chunk) {
            FragBF a, b;
            const uint4* ap =
                (const uint4*)(hB + arow * 256 + chunk * 32 + arow_off);
            a.u[0] = ap[0];
            a.u[1] = ap[2];
            int base = (((tile * 8 + chunk) * 32) + lane) << 1;
            b.u[0] = w2frag[base];
            b.u[1] = w2frag[base + 1];
            acc = __builtin_amdgcn_wmma_f32_16x16x32_bf16(
                false, a.v, false, b.v, (short)0, acc, false, false);
        }
        float wc = w2fp[(size_t)256 * 128 + tile * 16 + ncol];
        float s = 0.f;
#pragma unroll
        for (int r = 0; r < 8; ++r) {
            int row = (lane < 16) ? r : (r + 8);
            s += acc[r] + hcol[row] * wc;
        }
        msum[tile * 32 + lane] = s;   // partial sum over this lane's 8 rows
    }
    __syncthreads();

    // ---- complete neighbor sum per output column
    {
        int tile = tid >> 4, ln = tid & 15;
        colsum[tid] = msum[tile * 32 + ln] + msum[tile * 32 + ln + 16];
    }
    __syncthreads();

    // ---- group norm (2 groups of 64), leaky, residual
    if (tid < 2) {
        float mu = 0.f, m2 = 0.f;
        for (int i = 0; i < 64; ++i) {
            float x = colsum[tid * 64 + i];
            mu += x; m2 += x * x;
        }
        mu *= (1.f / 64.f);
        m2 *= (1.f / 64.f);
        gstat[tid * 2]     = mu;
        gstat[tid * 2 + 1] = rsqrtf(m2 - mu * mu + 1e-5f);
    }
    __syncthreads();
    {
        int g = tid >> 6;
        float xn = (colsum[tid] - gstat[g * 2]) * gstat[g * 2 + 1];
        float v  = leakyf(xn * gnw[tid] + gnb[tid]);
        float o  = center[(size_t)p * 128 + tid] + v;
        outp[(size_t)p * 128 + tid] = o;
        outb[(size_t)p * 128 + tid] = f32_to_bf16(o);
    }
}

// ---------------------------------------------------------------------------
// Host-side orchestration
// ---------------------------------------------------------------------------
extern "C" void kernel_launch(void* const* d_in, const int* in_sizes, int n_in,
                              void* d_out, int out_size, void* d_ws, size_t ws_size,
                              hipStream_t stream) {
    const float* xyz        = (const float*)d_in[0];
    const float* atom_xyz   = (const float*)d_in[1];
    const float* atomtypes  = (const float*)d_in[2];
    const int*   batch      = (const int*)d_in[3];
    const int*   atom_batch = (const int*)d_in[4];
    const float* tt_w1 = (const float*)d_in[5];
    const float* tt_b1 = (const float*)d_in[6];
    const float* tt_w2 = (const float*)d_in[7];
    const float* tt_b2 = (const float*)d_in[8];
    const float* aa_w1  = (const float*)d_in[9];
    const float* aa_b1  = (const float*)d_in[10];
    const float* aa_w2  = (const float*)d_in[11];
    const float* aa_b2  = (const float*)d_in[12];
    const float* aa_gnw = (const float*)d_in[13];
    const float* aa_gnb = (const float*)d_in[14];
    const float* em_w1  = (const float*)d_in[15];
    const float* em_b1  = (const float*)d_in[16];
    const float* em_w2  = (const float*)d_in[17];
    const float* em_b2  = (const float*)d_in[18];
    const float* em_gnw = (const float*)d_in[19];
    const float* em_gnb = (const float*)d_in[20];
    float* out = (float*)d_out;

    const int N_ = in_sizes[0] / 3;   // 16384
    const int M_ = in_sizes[1] / 3;   // 8192
    const int H = 257, Df = 128, NLy = 3;

    // ---- workspace carve-up
    char* ws = (char*)d_ws;
    size_t off = 0;
    auto carve = [&](size_t bytes) {
        size_t o = off;
        off += (bytes + 255) & ~(size_t)255;
        return (void*)(ws + o);
    };
    const size_t w1frag_l = 16 * 8 * 32 * 16;  // bf16 elems per layer (65536)
    const size_t w2frag_l = 8 * 8 * 32 * 16;   // 32768
    unsigned short* aaW1f = (unsigned short*)carve(NLy * w1frag_l * 2);
    unsigned short* emW1f = (unsigned short*)carve(NLy * w1frag_l * 2);
    unsigned short* aaW2f = (unsigned short*)carve(NLy * w2frag_l * 2);
    unsigned short* emW2f = (unsigned short*)carve(NLy * w2frag_l * 2);
    float* feat0 = (float*)carve((size_t)M_ * Df * 4);
    float* feat1 = (float*)carve((size_t)M_ * Df * 4);
    float* pe0   = (float*)carve((size_t)N_ * Df * 4);
    float* pe1   = (float*)carve((size_t)N_ * Df * 4);
    unsigned short* featB0 = (unsigned short*)carve((size_t)M_ * Df * 2);
    unsigned short* featB1 = (unsigned short*)carve((size_t)M_ * Df * 2);
    unsigned short* peB0   = (unsigned short*)carve((size_t)N_ * Df * 2);
    unsigned short* peB1   = (unsigned short*)carve((size_t)N_ * Df * 2);
    int*   kidx  = (int*)carve((size_t)M_ * 16 * 4);
    float* kd    = (float*)carve((size_t)M_ * 16 * 4);
    int*   kidx2 = (int*)carve((size_t)N_ * 16 * 4);
    float* kd2   = (float*)carve((size_t)N_ * 16 * 4);
    (void)ws_size; (void)n_in; (void)out_size;

    // ---- weight fragment prep (bf16, WMMA-B layout)
    prep_wfrag<<<512, 256, 0, stream>>>(aa_w1, aaW1f, 16, H, NLy, H * H);
    prep_wfrag<<<512, 256, 0, stream>>>(em_w1, emW1f, 16, H, NLy, H * H);
    prep_wfrag<<<256, 256, 0, stream>>>(aa_w2, aaW2f, 8, Df, NLy, H * Df);
    prep_wfrag<<<256, 256, 0, stream>>>(em_w2, emW2f, 8, Df, NLy, H * Df);

    // ---- atom-type embedding and pe init (fp32 + bf16 mirrors)
    atom_embed<<<(M_ + 127) / 128, 128, 0, stream>>>(
        atomtypes, tt_w1, tt_b1, tt_w2, tt_b2, feat0, featB0, M_);
    fill_ones_pair<<<512, 256, 0, stream>>>(pe0, peB0, N_ * Df);

    // ---- KNN (atoms->atoms, drop self; points->atoms); 16 queries/block
    knn_kernel<17, 1><<<(M_ + 15) / 16, 128, 0, stream>>>(
        atom_xyz, atom_xyz, atom_batch, atom_batch, M_, M_, kidx, kd);
    knn_kernel<16, 0><<<(N_ + 15) / 16, 128, 0, stream>>>(
        xyz, atom_xyz, batch, atom_batch, N_, M_, kidx2, kd2);

    // ---- atom message-passing layers (ping-pong feat0/feat1)
    const float*          a_in[3]   = {feat0, feat1, feat0};
    const unsigned short* aB_in[3]  = {featB0, featB1, featB0};
    float*                a_out[3]  = {feat1, feat0, feat1};
    unsigned short*       aB_out[3] = {featB1, featB0, featB1};
    for (int l = 0; l < NLy; ++l) {
        mp_layer<<<M_, 128, 0, stream>>>(
            M_, a_in[l], aB_in[l], aB_in[l], kidx, kd,
            (const uint4*)(aaW1f + (size_t)l * w1frag_l),
            (const uint4*)(aaW2f + (size_t)l * w2frag_l),
            aa_w1 + (size_t)l * H * H, aa_w2 + (size_t)l * H * Df,
            aa_b1 + (size_t)l * H, aa_b2 + (size_t)l * Df,
            aa_gnw + (size_t)l * Df, aa_gnb + (size_t)l * Df,
            a_out[l], aB_out[l]);
    }
    const unsigned short* atom_outB = featB1;   // layer 2 wrote feat1/featB1

    // ---- point message-passing layers (neighbors always read atom_out)
    const float*          p_in[3]   = {pe0, pe1, pe0};
    const unsigned short* pB_in[3]  = {peB0, peB1, peB0};
    float*                p_out[3]  = {pe1, pe0, out};
    unsigned short*       pB_out[3] = {peB1, peB0, peB1};
    for (int l = 0; l < NLy; ++l) {
        mp_layer<<<N_, 128, 0, stream>>>(
            N_, p_in[l], pB_in[l], atom_outB, kidx2, kd2,
            (const uint4*)(emW1f + (size_t)l * w1frag_l),
            (const uint4*)(emW2f + (size_t)l * w2frag_l),
            em_w1 + (size_t)l * H * H, em_w2 + (size_t)l * H * Df,
            em_b1 + (size_t)l * H, em_b2 + (size_t)l * Df,
            em_gnw + (size_t)l * Df, em_gnb + (size_t)l * Df,
            p_out[l], pB_out[l]);
    }
}
